// PolicyNetwork_89103391523227
// MI455X (gfx1250) — matrix operand order repaired and checked by reading, not compiled, not run on hardware
//
#include <hip/hip_runtime.h>
#include <hip/hip_bf16.h>
#include <stdint.h>

#define H     512
#define OPS   12
#define E     112
#define FOURH 2048
#define KCAT  1024

typedef __attribute__((ext_vector_type(16))) _Float16 v16h;
typedef __attribute__((ext_vector_type(8)))  float    v8f;
typedef uint32_t u32x4 __attribute__((ext_vector_type(4)));
typedef uint32_t u32x8 __attribute__((ext_vector_type(8)));

union V16H {
    v16h     v;
    _Float16 h[16];
    u32x4    q[2];
};

__device__ __forceinline__ float sigm(float x) {
    return 1.0f / (1.0f + __expf(-x));
}

__device__ __forceinline__ float u01hash(uint32_t x) {
    x ^= x >> 17; x *= 0xed5ad4bbu;
    x ^= x >> 11; x *= 0xac4c1b51u;
    x ^= x >> 15; x *= 0x31848babu;
    x ^= x >> 14;
    return (float)(x >> 8) * (1.0f / 16777216.0f) + 1.0e-7f;
}

// ---------------------------------------------------------------------------
// One-time precompute: pack [Wih | Whh] -> f16 (2048 x 1024, row-major),
// emb -> f16, bias sums -> f32.  Weights then live in L2 (192MB) as f16.
// ---------------------------------------------------------------------------
__global__ void convert_kernel(const float* __restrict__ Wih0, const float* __restrict__ Whh0,
                               const float* __restrict__ bih0, const float* __restrict__ bhh0,
                               const float* __restrict__ Wih1, const float* __restrict__ Whh1,
                               const float* __restrict__ bih1, const float* __restrict__ bhh1,
                               const float* __restrict__ emb,
                               _Float16* __restrict__ Wcat0, _Float16* __restrict__ Wcat1,
                               _Float16* __restrict__ embh,
                               float* __restrict__ bsum0, float* __restrict__ bsum1) {
    const int stride = gridDim.x * blockDim.x;
    const int tid0   = blockIdx.x * blockDim.x + threadIdx.x;
    const int total  = FOURH * KCAT;
    for (int idx = tid0; idx < total; idx += stride) {
        const int r = idx >> 10;
        const int c = idx & (KCAT - 1);
        float v0, v1;
        if (c < H) { v0 = Wih0[r * H + c];       v1 = Wih1[r * H + c]; }
        else       { v0 = Whh0[r * H + (c - H)]; v1 = Whh1[r * H + (c - H)]; }
        Wcat0[idx] = (_Float16)v0;
        Wcat1[idx] = (_Float16)v1;
    }
    if (tid0 < OPS * H) embh[tid0] = (_Float16)emb[tid0];
    if (tid0 < FOURH) {
        bsum0[tid0] = bih0[tid0] + bhh0[tid0];
        bsum1[tid0] = bih1[tid0] + bhh1[tid0];
    }
}

// ---------------------------------------------------------------------------
// gates[0..2047] = Wcat(2048x1024,f16) * vcat(1024,f16) + bsum
// One A-fragment buffer reused across the 4 M-tiles (keeps VGPR count under
// the 8-waves/SIMD cap; with 32 waves resident, latency is hidden by wave
// switching, not in-wave pipelining).  B (ds b128 x2, broadcast + cndmask)
// is shared by 4 WMMAs per k-step.
// ---------------------------------------------------------------------------
__device__ __forceinline__ void matvec2048(const _Float16* __restrict__ W,
                                           const float* __restrict__ bsum,
                                           const _Float16* __restrict__ vcat,
                                           float* __restrict__ gatesOut,
                                           int wave, int lane) {
    const int n    = lane & 15;
    const int half = lane >> 4;
    const u32x4 zero4 = {0u, 0u, 0u, 0u};

    const int row0 = wave * 64 + n;
    const _Float16* r0 = W + (size_t)(row0)      * KCAT;
    const _Float16* r1 = W + (size_t)(row0 + 16) * KCAT;
    const _Float16* r2 = W + (size_t)(row0 + 32) * KCAT;
    const _Float16* r3 = W + (size_t)(row0 + 48) * KCAT;

    v8f acc0 = {}, acc1 = {}, acc2 = {}, acc3 = {};

#pragma clang loop unroll(disable)
    for (int kb = 0; kb < KCAT; kb += 32) {
        // B: 16 consecutive f16 of the vector per lane-half, column N=0 only
        V16H b;
        const int kbB = kb + half * 16;
        b.q[0] = *(const u32x4*)(vcat + kbB);
        b.q[1] = *(const u32x4*)(vcat + kbB + 8);
        if (n != 0) { b.q[0] = zero4; b.q[1] = zero4; }   // v_cndmask, no exec ops

        const int ka0 = kb + half * 16;       // regs 0-3: K = half*16 .. +7
        const int ka1 = kb + half * 16 + 8;   // regs 4-7: K = half*16+8 .. +7
        V16H a;
        a.q[0] = *(const u32x4*)(r0 + ka0);
        a.q[1] = *(const u32x4*)(r0 + ka1);
        acc0 = __builtin_amdgcn_wmma_f32_16x16x32_f16(false, a.v, false, b.v, (short)0, acc0, false, false);
        a.q[0] = *(const u32x4*)(r1 + ka0);
        a.q[1] = *(const u32x4*)(r1 + ka1);
        acc1 = __builtin_amdgcn_wmma_f32_16x16x32_f16(false, a.v, false, b.v, (short)0, acc1, false, false);
        a.q[0] = *(const u32x4*)(r2 + ka0);
        a.q[1] = *(const u32x4*)(r2 + ka1);
        acc2 = __builtin_amdgcn_wmma_f32_16x16x32_f16(false, a.v, false, b.v, (short)0, acc2, false, false);
        a.q[0] = *(const u32x4*)(r3 + ka0);
        a.q[1] = *(const u32x4*)(r3 + ka1);
        acc3 = __builtin_amdgcn_wmma_f32_16x16x32_f16(false, a.v, false, b.v, (short)0, acc3, false, false);
    }

    if (n == 0) {   // D column N=0: lane0 -> M=j, lane16 -> M=8+j
        const int base = wave * 64 + half * 8;
#pragma unroll
        for (int j = 0; j < 8; ++j) gatesOut[base      + j] = acc0[j] + bsum[base      + j];
#pragma unroll
        for (int j = 0; j < 8; ++j) gatesOut[base + 16 + j] = acc1[j] + bsum[base + 16 + j];
#pragma unroll
        for (int j = 0; j < 8; ++j) gatesOut[base + 32 + j] = acc2[j] + bsum[base + 32 + j];
#pragma unroll
        for (int j = 0; j < 8; ++j) gatesOut[base + 48 + j] = acc3[j] + bsum[base + 48 + j];
    }
}

// ---------------------------------------------------------------------------
// hmlp[n] = relu( sum_k h1[k] * W1[k*H + n] + b1[n] )
// Column-major cW1[e] tiles are staged by the Tensor Data Mover into
// double-buffered per-wave LDS slabs.  TDM transfers complete in order per
// wave, so "issue next tile; s_wait_tensorcnt 1" overlaps the DMA for kb+32
// with the convert+WMMA for kb.  Double buffering costs LDS, not VGPRs.
// ---------------------------------------------------------------------------
__device__ __forceinline__ void mlp512_tdm(const float* __restrict__ W1,
                                           const float* __restrict__ b1,
                                           const _Float16* __restrict__ h1h,
                                           float* __restrict__ hmlpOut,
                                           const float* __restrict__ slabF,   // this wave's 2 x [32][16] slabs
                                           uint32_t slabLdsByte,              // LDS byte address of slab 0
                                           int wave, int lane) {
    const int n    = lane & 15;
    const int half = lane >> 4;
    const u32x4 zero4 = {0u, 0u, 0u, 0u};

    const uint64_t base = (uint64_t)(uintptr_t)W1 + (uint64_t)(wave * 16) * 4ull;

    auto issueTDM = [&](int kb, uint32_t ldsByte) {
        const uint64_t gaddr = base + (uint64_t)kb * (uint64_t)(H * 4);
        u32x4 g0;
        g0[0] = 1u;                                        // count=1, user mode
        g0[1] = ldsByte;                                   // lds_addr (bytes)
        g0[2] = (uint32_t)gaddr;                           // global_addr[31:0]
        g0[3] = (uint32_t)(gaddr >> 32) | (2u << 30);      // addr[56:32] | type=2
        u32x8 g1;
        g1[0] = 0x20000u;                                  // data_size=4B, no mcast
        g1[1] = (uint32_t)H << 16;                         // tensor_dim0 lo16 = 512
        g1[2] = (uint32_t)H << 16;                         // tensor_dim1 lo16 = 512
        g1[3] = 16u << 16;                                 // tile_dim0 = 16
        g1[4] = 32u;                                       // tile_dim1 = 32
        g1[5] = (uint32_t)H;                               // tensor_dim0_stride = 512
        g1[6] = 0u;
        g1[7] = 0u;
        asm volatile("tensor_load_to_lds %0, %1" :: "s"(g0), "s"(g1) : "memory");
    };

    v8f acc = {};
    int cur = 0;
    issueTDM(0, slabLdsByte);                              // prologue

#pragma clang loop unroll(disable)
    for (int kb = 0; kb < H; kb += 32) {
        if (kb + 32 < H) {
            issueTDM(kb + 32, slabLdsByte + (uint32_t)((cur ^ 1) * 32 * 16 * 4));
            __builtin_amdgcn_s_wait_tensorcnt(1);          // older transfer done
        } else {
            __builtin_amdgcn_s_wait_tensorcnt(0);          // last transfer done
        }
        const float* tileF = slabF + cur * (32 * 16);

        // ---- A fragment: a[m,k] = (f16)tile[k][m] ----
        V16H a;
#pragma unroll
        for (int i = 0; i < 8; ++i) {
            const int k0 = half * 16 + 2 * i;
            a.h[2 * i]     = (_Float16)tileF[(k0    ) * 16 + n];
            a.h[2 * i + 1] = (_Float16)tileF[(k0 + 1) * 16 + n];
        }
        // ---- B fragment: h1 vector in column N=0 ----
        V16H b;
        const int kbB = kb + half * 16;
        b.q[0] = *(const u32x4*)(h1h + kbB);
        b.q[1] = *(const u32x4*)(h1h + kbB + 8);
        if (n != 0) { b.q[0] = zero4; b.q[1] = zero4; }

        acc = __builtin_amdgcn_wmma_f32_16x16x32_f16(false, a.v, false, b.v, (short)0, acc, false, false);
        cur ^= 1;
    }

    if (n == 0) {
#pragma unroll
        for (int j = 0; j < 8; ++j) {
            const int nn = wave * 16 + half * 8 + j;
            const float v = acc[j] + b1[nn];
            hmlpOut[nn] = v > 0.0f ? v : 0.0f;
        }
    }
}

// ---------------------------------------------------------------------------
// Persistent single-workgroup scan over all 112 steps (32 wave32 waves).
// ---------------------------------------------------------------------------
__global__ void __launch_bounds__(1024)
policy_kernel(const int* __restrict__ tempI,
              const float* __restrict__ h0, const float* __restrict__ c0,
              const _Float16* __restrict__ Wcat0, const float* __restrict__ bsum0,
              const _Float16* __restrict__ Wcat1, const float* __restrict__ bsum1,
              const _Float16* __restrict__ embh,
              const float* __restrict__ cW1, const float* __restrict__ cb1,
              const float* __restrict__ cW2, const float* __restrict__ cb2,
              float* __restrict__ out) {
    __shared__ float    hS[2][H];
    __shared__ float    cS[2][H];
    __shared__ float    gatesS[FOURH];
    __shared__ _Float16 vcat[KCAT];
    __shared__ _Float16 h1h[H];
    __shared__ float    hmlpS[H];
    __shared__ float    logitsS[OPS];
    __shared__ int      prevA;
    __shared__ float    tempS;
    __shared__ float    w1tileS[32][2][32 * 16];   // per-wave double-buffered TDM slabs (128KB)

    const int tid  = threadIdx.x;
    const int wave = tid >> 5;
    const int lane = tid & 31;

    // low 32 bits of the generic pointer are the LDS byte offset (ISA 10.2)
    const uint32_t slabLdsByte = (uint32_t)(uintptr_t)(const void*)&w1tileS[wave][0][0];

    if (tid < H) {
        hS[0][tid] = h0[tid];       hS[1][tid] = h0[H + tid];
        cS[0][tid] = c0[tid];       cS[1][tid] = c0[H + tid];
    }
    if (tid == 0) {
        prevA = 0;
        const int   iv = tempI[0];
        const float fv = __int_as_float(iv);
        tempS = (iv >= 1 && iv <= 1000000) ? (float)iv
              : ((fv > 1e-6f && fv < 1e6f) ? fv : 1.0f);
    }
    __syncthreads();

    for (int e = 0; e < E; ++e) {
        // ---- build [x ; h0_prev] in f16 ----
        if (tid < H) {
            const float xv = (e == 0) ? 0.0f : (float)embh[prevA * H + tid];
            vcat[tid]     = (_Float16)xv;
            vcat[H + tid] = (_Float16)hS[0][tid];
        }
        if (tid == 0 && e + 1 < E)
            __builtin_prefetch(cW1 + (size_t)(e + 1) * H * H, 0, 1); // global_prefetch_b8
        __syncthreads();

        matvec2048(Wcat0, bsum0, vcat, gatesS, wave, lane);
        __syncthreads();

        if (tid < H) {  // LSTM cell 0 (gate order i,f,g,o)
            const float ig = sigm(gatesS[tid]);
            const float fg = sigm(gatesS[H + tid]);
            const float gg = tanhf(gatesS[2 * H + tid]);
            const float og = sigm(gatesS[3 * H + tid]);
            const float c2 = fg * cS[0][tid] + ig * gg;
            const float h2 = og * tanhf(c2);
            cS[0][tid] = c2;
            hS[0][tid] = h2;
        }
        __syncthreads();

        if (tid < H) {  // [h0_new ; h1_prev]
            vcat[tid]     = (_Float16)hS[0][tid];
            vcat[H + tid] = (_Float16)hS[1][tid];
        }
        __syncthreads();

        matvec2048(Wcat1, bsum1, vcat, gatesS, wave, lane);
        __syncthreads();

        if (tid < H) {  // LSTM cell 1
            const float ig = sigm(gatesS[tid]);
            const float fg = sigm(gatesS[H + tid]);
            const float gg = tanhf(gatesS[2 * H + tid]);
            const float og = sigm(gatesS[3 * H + tid]);
            const float c2 = fg * cS[1][tid] + ig * gg;
            const float h2 = og * tanhf(c2);
            cS[1][tid] = c2;
            hS[1][tid] = h2;
            h1h[tid]   = (_Float16)h2;
        }
        __syncthreads();

        // ---- controller MLP hidden via pipelined TDM staging of cW1[e] ----
        mlp512_tdm(cW1 + (size_t)e * H * H, cb1 + (size_t)e * H, h1h, hmlpS,
                   &w1tileS[wave][0][0], slabLdsByte, wave, lane);
        __syncthreads();

        // ---- logits: 12 dot products, one wave each, shfl reduction ----
        if (wave < OPS) {
            float s = 0.0f;
            const float* W2 = cW2 + (size_t)e * H * OPS + wave;
            for (int k = lane; k < H; k += 32)
                s += hmlpS[k] * W2[(size_t)k * OPS];
#pragma unroll
            for (int off = 16; off > 0; off >>= 1)
                s += __shfl_down(s, off, 32);
            if (lane == 0)
                logitsS[wave] = s + cb2[(size_t)e * OPS + wave];
        }
        __syncthreads();

        // ---- softmax / gumbel-argmax / entropy ----
        if (tid == 0) {
            const float t = tempS;
            float l[OPS];
            float mx = -3.4e38f;
            for (int o = 0; o < OPS; ++o) {
                l[o] = logitsS[o] / t;
                mx = fmaxf(mx, l[o]);
            }
            float se = 0.0f;
            for (int o = 0; o < OPS; ++o) se += __expf(l[o] - mx);
            const float logZ = mx + __logf(se);

            int best = 0; float bestv = -3.4e38f;
            for (int o = 0; o < OPS; ++o) {
                const float u = u01hash((uint32_t)(e * OPS + o) * 2654435761u + 42u);
                const float g = -__logf(-__logf(u));
                const float v = l[o] + g;
                if (v > bestv) { bestv = v; best = o; }
            }
            float ent = 0.0f;
            for (int o = 0; o < OPS; ++o) {
                const float lpo = l[o] - logZ;
                ent -= __expf(lpo) * lpo;
            }
            out[e]         = (float)best;
            out[E + e]     = l[best] - logZ;
            out[2 * E + e] = ent;
            prevA = best;
        }
        __syncthreads();
    }
}

extern "C" void kernel_launch(void* const* d_in, const int* in_sizes, int n_in,
                              void* d_out, int out_size, void* d_ws, size_t ws_size,
                              hipStream_t stream) {
    (void)in_sizes; (void)n_in; (void)out_size; (void)ws_size;

    const int*   tempI = (const int*)d_in[0];
    const float* h0    = (const float*)d_in[1];
    const float* c0    = (const float*)d_in[2];
    const float* Wih0  = (const float*)d_in[3];
    const float* Whh0  = (const float*)d_in[4];
    const float* bih0  = (const float*)d_in[5];
    const float* bhh0  = (const float*)d_in[6];
    const float* Wih1  = (const float*)d_in[7];
    const float* Whh1  = (const float*)d_in[8];
    const float* bih1  = (const float*)d_in[9];
    const float* bhh1  = (const float*)d_in[10];
    const float* emb   = (const float*)d_in[11];
    const float* cW1   = (const float*)d_in[12];
    const float* cb1   = (const float*)d_in[13];
    const float* cW2   = (const float*)d_in[14];
    const float* cb2   = (const float*)d_in[15];
    float*       out   = (float*)d_out;

    char* ws = (char*)d_ws;
    _Float16* Wcat0 = (_Float16*)ws;                         // 4 MiB
    _Float16* Wcat1 = (_Float16*)(ws + 4194304);             // 4 MiB
    _Float16* embh  = (_Float16*)(ws + 8388608);             // 12 KiB
    float*    bsum0 = (float*)(ws + 8388608 + 12288);        // 8 KiB
    float*    bsum1 = bsum0 + FOURH;                         // 8 KiB

    convert_kernel<<<1024, 256, 0, stream>>>(Wih0, Whh0, bih0, bhh0,
                                             Wih1, Whh1, bih1, bhh1,
                                             emb, Wcat0, Wcat1, embh, bsum0, bsum1);

    policy_kernel<<<1, 1024, 0, stream>>>(tempI, h0, c0,
                                          Wcat0, bsum0, Wcat1, bsum1, embh,
                                          cW1, cb1, cW2, cb2, out);
}
